// SGHFM_66803921322219
// MI455X (gfx1250) — matrix operand order repaired and checked by reading, not compiled, not run on hardware
//
#include <hip/hip_runtime.h>
#include <hip/hip_bf16.h>
#include <math.h>

typedef __attribute__((ext_vector_type(16))) _Float16 v16h;
typedef __attribute__((ext_vector_type(8)))  _Float16 v8h;
typedef __attribute__((ext_vector_type(8)))  float    v8f;

#define BB 2
#define TT 1024
#define DD 256
#define NN 1024
#define HH 8
#define HDIM 128
#define KW 32
#define FFD 1024          // 4*D
#define VOC 32000
#define MM (BB*TT)        // 2048
#define TWO_PI_F 6.28318530717958647692f

// ---------------------------------------------------------------------------
// Reductions
// ---------------------------------------------------------------------------
__device__ inline float blockReduceSum(float v, float* lds, int nwaves) {
    #pragma unroll
    for (int off = 16; off; off >>= 1) v += __shfl_down(v, off, 32);
    int lane = threadIdx.x & 31;
    int w    = threadIdx.x >> 5;
    __syncthreads();
    if (lane == 0) lds[w] = v;
    __syncthreads();
    float r = 0.f;
    for (int i = 0; i < nwaves; ++i) r += lds[i];
    return r;
}

__device__ inline float gelu_exact(float x) {
    return 0.5f * x * (1.f + erff(x * 0.70710678118654752440f));
}
__device__ inline float softplus_f(float x) {
    return fmaxf(x, 0.f) + log1pf(expf(-fabsf(x)));
}

// ---------------------------------------------------------------------------
// Weight conversion kernels (fp32 -> f16), TRANSPOSED to [N][K] for the GEMM
// B-operand so each lane's WMMA fragment is two contiguous 16-byte runs along K.
// ---------------------------------------------------------------------------
// in [R][C] f32 -> out [C][R] f16
__global__ void cvt_transpose_f16(const float* __restrict__ in, _Float16* __restrict__ out,
                                  int R, int C) {
    int i = blockIdx.x * 256 + threadIdx.x;
    if (i >= R * C) return;
    int r = i / C, c = i % C;
    out[(long long)c * R + r] = (_Float16)in[i];
}

// Wq/Wv [H,D,HD] -> Bt [(h*HD+e)][d]  (i.e. [N][K=D])
__global__ void cvt_wqv_t(const float* __restrict__ w, _Float16* __restrict__ out) {
    int i = blockIdx.x * 256 + threadIdx.x;     // over H*D*HD
    if (i >= HH * DD * HDIM) return;
    int h = i / (DD * HDIM);
    int rem = i % (DD * HDIM);
    int d = rem / HDIM, e = rem % HDIM;
    out[((long long)(h * HDIM + e)) * DD + d] = (_Float16)w[i];
}

// ---------------------------------------------------------------------------
// Embedding + quantized RoPE
// ---------------------------------------------------------------------------
__global__ void embed_rope(const int* __restrict__ tokens, const float* __restrict__ emb,
                           const float* __restrict__ field, float* __restrict__ x) {
    int bt = blockIdx.x;          // b*T + t
    int d  = threadIdx.x;         // 0..255
    int t  = bt % TT;
    int tok = tokens[bt];
    float v  = emb[(long long)tok * DD + d] + field[d];
    int   dp = (d & 1) ? (d - 1) : (d + 1);
    float vp = emb[(long long)tok * DD + dp] + field[dp];
    float xrot = (d & 1) ? vp : -vp;                 // [-x_odd, x_even] interleave
    float expo = (float)((d >> 1) * 2) / (float)DD;
    float freq = exp2f(-16.f * expo) * (1.f / TWO_PI_F);
    float ph   = fmodf((float)t * freq, 1.f) * TWO_PI_F;
    x[(long long)bt * DD + d] = v * cosf(ph) + xrot * sinf(ph);
}

// ---------------------------------------------------------------------------
// LayerNorm (row = 256 elems) -> f16
// ---------------------------------------------------------------------------
__global__ void layernorm_f16(const float* __restrict__ x, const float* __restrict__ g,
                              const float* __restrict__ b, _Float16* __restrict__ y) {
    __shared__ float lds[8];
    int row = blockIdx.x;
    int d   = threadIdx.x;
    float v = x[(long long)row * DD + d];
    float s  = blockReduceSum(v, lds, 8);
    float mu = s * (1.f / DD);
    float df = v - mu;
    float s2 = blockReduceSum(df * df, lds, 8);
    float var = s2 * (1.f / DD);
    y[(long long)row * DD + d] = (_Float16)(df * rsqrtf(var + 1e-5f) * g[d] + b[d]);
}

// ---------------------------------------------------------------------------
// WMMA GEMM: out = act(A[M,K] @ B[K,N] + bias), B given TRANSPOSED as Bt[N][K].
// Wave tile 32x32 (4 accumulators, 2x reuse of each fragment), block 8 waves
// tiling 64(M) x 128(N). Requires M%64==0, N%128==0, K%32==0 (all hold here).
// All fragment loads are explicit aligned vector loads -> global_load_b128.
// act: 0=none 1=relu 2=gelu(exact)
// ---------------------------------------------------------------------------
union frag16 { v16h v; v8h h[2]; };

template<int KSZ>
__global__ void gemm_wmma(const _Float16* __restrict__ A, const _Float16* __restrict__ Bt,
                          const float* __restrict__ bias, float* __restrict__ o32,
                          _Float16* __restrict__ o16, int Nsz, int act) {
    int lane = threadIdx.x;                 // 0..31
    int wave = threadIdx.y;                 // 0..7
    int wy = wave >> 2, wx = wave & 3;
    int half = lane >> 4, lm = lane & 15;
    int rowBase = blockIdx.y * 64 + wy * 32;
    int colBase = blockIdx.x * 128 + wx * 32;

    const _Float16* A0 = A  + (long long)(rowBase + lm)      * KSZ;
    const _Float16* A1 = A  + (long long)(rowBase + 16 + lm) * KSZ;
    const _Float16* B0 = Bt + (long long)(colBase + lm)      * KSZ;
    const _Float16* B1 = Bt + (long long)(colBase + 16 + lm) * KSZ;

    v8f c00 = {}, c01 = {}, c10 = {}, c11 = {};

    #pragma unroll 4
    for (int k0 = 0; k0 < KSZ; k0 += 32) {
        if (k0 + 128 < KSZ) {               // prefetch next K-panel
            __builtin_prefetch(A0 + k0 + 128, 0, 1);
            __builtin_prefetch(B0 + k0 + 128, 0, 1);
        }
        // A frag: two 16-byte-aligned 8-half runs per lane
        frag16 a0, a1, b0, b1;
        a0.h[0] = *(const v8h*)(A0 + k0 + half * 8);
        a0.h[1] = *(const v8h*)(A0 + k0 + 16 + half * 8);
        a1.h[0] = *(const v8h*)(A1 + k0 + half * 8);
        a1.h[1] = *(const v8h*)(A1 + k0 + 16 + half * 8);
        // B frag: two 16-byte-aligned 8-half runs per lane
        b0.h[0] = *(const v8h*)(B0 + k0 + half * 8);
        b0.h[1] = *(const v8h*)(B0 + k0 + 16 + half * 8);
        b1.h[0] = *(const v8h*)(B1 + k0 + half * 8);
        b1.h[1] = *(const v8h*)(B1 + k0 + 16 + half * 8);

        c00 = __builtin_amdgcn_wmma_f32_16x16x32_f16(false, a0.v, false, b0.v, (short)0, c00, false, false);
        c01 = __builtin_amdgcn_wmma_f32_16x16x32_f16(false, a0.v, false, b1.v, (short)0, c01, false, false);
        c10 = __builtin_amdgcn_wmma_f32_16x16x32_f16(false, a1.v, false, b0.v, (short)0, c10, false, false);
        c11 = __builtin_amdgcn_wmma_f32_16x16x32_f16(false, a1.v, false, b1.v, (short)0, c11, false, false);
    }

    int col0 = colBase + lm, col1 = colBase + 16 + lm;
    float bia0 = bias ? bias[col0] : 0.f;
    float bia1 = bias ? bias[col1] : 0.f;

    #pragma unroll
    for (int r = 0; r < 8; ++r) {
        int row0 = rowBase + half * 8 + r;        // C/D layout: M = r + 8*(lane>>4)
        int row1 = row0 + 16;
        float v00 = c00[r] + bia0, v01 = c01[r] + bia1;
        float v10 = c10[r] + bia0, v11 = c11[r] + bia1;
        if (act == 1) {
            v00 = fmaxf(v00, 0.f); v01 = fmaxf(v01, 0.f);
            v10 = fmaxf(v10, 0.f); v11 = fmaxf(v11, 0.f);
        } else if (act == 2) {
            v00 = gelu_exact(v00); v01 = gelu_exact(v01);
            v10 = gelu_exact(v10); v11 = gelu_exact(v11);
        }
        long long i00 = (long long)row0 * Nsz + col0;
        long long i01 = (long long)row0 * Nsz + col1;
        long long i10 = (long long)row1 * Nsz + col0;
        long long i11 = (long long)row1 * Nsz + col1;
        if (o32) { o32[i00] = v00; o32[i01] = v01; o32[i10] = v10; o32[i11] = v11; }
        if (o16) { o16[i00] = (_Float16)v00; o16[i01] = (_Float16)v01;
                   o16[i10] = (_Float16)v10; o16[i11] = (_Float16)v11; }
    }
}

// ---------------------------------------------------------------------------
// Field signature: sig[b,n] = mean_t qs + 0.5*mean_{t%8==0} qs
// ---------------------------------------------------------------------------
__global__ void sig_kernel(const float* __restrict__ qs, float* __restrict__ sig) {
    int i = blockIdx.x * 256 + threadIdx.x;   // B*NN
    if (i >= BB * NN) return;
    int b = i / NN, n = i % NN;
    float s1 = 0.f, s2 = 0.f;
    for (int t = 0; t < TT; ++t) {
        float q = qs[((long long)b * TT + t) * NN + n];
        s1 += q;
        if ((t & 7) == 0) s2 += q;
    }
    sig[i] = s1 * (1.f / TT) + s2 * (1.f / (TT / 8)) * 0.5f;
}

// metric MLP layer 1: hid[b,j] = gelu(sig[b,:] @ mW1[:,j] + mb1[j])   (j < 512)
__global__ void metric_l1(const float* __restrict__ sig, const float* __restrict__ mW1,
                          const float* __restrict__ mb1, float* __restrict__ hid) {
    int i = blockIdx.x * 256 + threadIdx.x;
    if (i >= BB * (NN / 2)) return;
    int b = i / (NN / 2), j = i % (NN / 2);
    float s = mb1[j];
    for (int n = 0; n < NN; ++n) s += sig[b * NN + n] * mW1[(long long)n * (NN / 2) + j];
    hid[i] = gelu_exact(s);
}

// layer 2 + softplus metric: metric[b,n] = softplus(base[n] + 0.1*mod)+1e-6
__global__ void metric_l2(const float* __restrict__ hid, const float* __restrict__ mW2,
                          const float* __restrict__ mb2, const float* __restrict__ base,
                          float* __restrict__ metric) {
    int i = blockIdx.x * 256 + threadIdx.x;
    if (i >= BB * NN) return;
    int b = i / NN, n = i % NN;
    float s = mb2[n];
    for (int j = 0; j < NN / 2; ++j) s += hid[b * (NN / 2) + j] * mW2[(long long)j * NN + n];
    metric[i] = softplus_f(base[n] + 0.1f * s) + 1e-6f;
}

// ---------------------------------------------------------------------------
// Causal K-window metric attention + Hebbian + multiplicative gate.
// One 128-thread block per (b,t,h). Outputs gated f16 [B,T,N].
// ---------------------------------------------------------------------------
__global__ void window_attn(const float* __restrict__ qs, const float* __restrict__ vs,
                            const float* __restrict__ metric, const float* __restrict__ hebb,
                            _Float16* __restrict__ gated) {
    __shared__ float lds[4];
    __shared__ float qsh[HDIM];
    int blk = blockIdx.x;                       // (b*T + t)*H + h
    int h = blk % HH;
    int t = (blk / HH) % TT;
    int b = blk / (HH * TT);
    int d = threadIdx.x;                        // 0..127

    long long base = ((long long)(b * TT + t) * HH + h) * HDIM;
    float q = qs[base + d];
    float m = metric[b * NN + h * HDIM + d];

    float acc = 0.f, wsum = 0.f;
    for (int k = 0; k < KW; ++k) {
        int j = t - (KW - 1) + k;               // padded window incl. zeros for j<0
        float qw = 0.f, vw = 0.f;
        if (j >= 0) {
            long long jb = ((long long)(b * TT + j) * HH + h) * HDIM;
            qw = qs[jb + d];
            vw = vs[jb + d];
        }
        float df  = q - qw;
        float tot = blockReduceSum(m * df * df, lds, 4);
        float w   = expf(-sqrtf(tot + 1e-8f));
        wsum += w;
        acc  += w * vw;
    }
    float ctx = acc / (wsum + 1e-8f);

    // Hebbian: qn = q / max(||q||, 1e-12); hctx_e = sum_d qn_d * hebb[h,d,e]
    float nq = blockReduceSum(q * q, lds, 4);
    float qn = q / fmaxf(sqrtf(nq), 1e-12f);
    __syncthreads();
    qsh[d] = qn;
    __syncthreads();
    const float* hb = hebb + (long long)h * HDIM * HDIM;
    float hc = 0.f;
    for (int dd = 0; dd < HDIM; ++dd) hc += qsh[dd] * hb[(long long)dd * HDIM + d];

    gated[base + d] = (_Float16)(q * (ctx + hc));
}

// x += a + b (residual)
__global__ void residual_add(float* __restrict__ x, const float* __restrict__ a,
                             const float* __restrict__ b, int n) {
    int i = blockIdx.x * 256 + threadIdx.x;
    if (i < n) x[i] += a[i] + b[i];
}

// ---------------------------------------------------------------------------
// Host launcher
// ---------------------------------------------------------------------------
template<int KSZ>
static inline void launch_gemm(const _Float16* A, const _Float16* Bt, const float* bias,
                               float* o32, _Float16* o16, int Msz, int Nsz, int act,
                               hipStream_t s) {
    dim3 grid(Nsz / 128, Msz / 64);
    dim3 blk(32, 8);
    gemm_wmma<KSZ><<<grid, blk, 0, s>>>(A, Bt, bias, o32, o16, Nsz, act);
}

static inline size_t alignup(size_t v) { return (v + 255) & ~(size_t)255; }

extern "C" void kernel_launch(void* const* d_in, const int* in_sizes, int n_in,
                              void* d_out, int out_size, void* d_ws, size_t ws_size,
                              hipStream_t stream) {
    const int*   tokens   = (const int*)  d_in[0];
    const float* emb      = (const float*)d_in[1];
    const float* field    = (const float*)d_in[2];
    const float* Wq       = (const float*)d_in[3];
    const float* Wv       = (const float*)d_in[4];
    const float* mW1      = (const float*)d_in[5];
    const float* mb1      = (const float*)d_in[6];
    const float* mW2      = (const float*)d_in[7];
    const float* mb2      = (const float*)d_in[8];
    const float* base_m   = (const float*)d_in[9];
    const float* decoder  = (const float*)d_in[10];
    const float* hebbian  = (const float*)d_in[11];
    const float* fW1      = (const float*)d_in[12];
    const float* fb1      = (const float*)d_in[13];
    const float* fW2      = (const float*)d_in[14];
    const float* fb2      = (const float*)d_in[15];
    const float* g1       = (const float*)d_in[16];
    const float* b1       = (const float*)d_in[17];
    const float* g2       = (const float*)d_in[18];
    const float* b2       = (const float*)d_in[19];
    const float* gf       = (const float*)d_in[20];
    const float* bf       = (const float*)d_in[21];
    const float* Wl       = (const float*)d_in[22];
    const float* bl       = (const float*)d_in[23];
    float* out = (float*)d_out;

    uint8_t* ws = (uint8_t*)d_ws;
    size_t off = 0;
    auto take = [&](size_t bytes) { uint8_t* p = ws + off; off += alignup(bytes); return p; };

    float*    x      = (float*)   take((size_t)MM * DD * 4);
    _Float16* yf16   = (_Float16*)take((size_t)MM * DD * 2);
    float*    qsB    = (float*)   take((size_t)MM * NN * 4);
    float*    vsB    = (float*)   take((size_t)MM * NN * 4);
    float*    sig    = (float*)   take((size_t)BB * NN * 4);
    float*    hid    = (float*)   take((size_t)BB * (NN / 2) * 4);
    float*    metric = (float*)   take((size_t)BB * NN * 4);
    _Float16* gated  = (_Float16*)take((size_t)MM * NN * 2);
    float*    decOut = (float*)   take((size_t)MM * DD * 4);
    _Float16* h1     = (_Float16*)take((size_t)MM * FFD * 2);
    float*    ffnOut = (float*)   take((size_t)MM * DD * 4);
    // weights, all f16 TRANSPOSED to [N][K]
    _Float16* WqT    = (_Float16*)take((size_t)NN * DD * 2);
    _Float16* WvT    = (_Float16*)take((size_t)NN * DD * 2);
    _Float16* decT   = (_Float16*)take((size_t)DD * NN * 2);
    _Float16* fW1T   = (_Float16*)take((size_t)FFD * DD * 2);
    _Float16* fW2T   = (_Float16*)take((size_t)DD * FFD * 2);
    _Float16* WlT    = (_Float16*)take((size_t)VOC * DD * 2);
    (void)ws_size; (void)n_in; (void)in_sizes; (void)out_size;

    // --- one-time (per launch) weight conversion to transposed f16 ---
    cvt_wqv_t<<<(HH * DD * HDIM + 255) / 256, 256, 0, stream>>>(Wq, WqT);
    cvt_wqv_t<<<(HH * DD * HDIM + 255) / 256, 256, 0, stream>>>(Wv, WvT);
    cvt_transpose_f16<<<(NN * DD + 255) / 256, 256, 0, stream>>>(decoder, decT, NN, DD);
    cvt_transpose_f16<<<(DD * FFD + 255) / 256, 256, 0, stream>>>(fW1, fW1T, DD, FFD);
    cvt_transpose_f16<<<(FFD * DD + 255) / 256, 256, 0, stream>>>(fW2, fW2T, FFD, DD);
    cvt_transpose_f16<<<(DD * VOC + 255) / 256, 256, 0, stream>>>(Wl, WlT, DD, VOC);

    // --- embedding + RoPE ---
    embed_rope<<<MM, DD, 0, stream>>>(tokens, emb, field, x);

    // --- 2 evolution steps ---
    for (int step = 0; step < 2; ++step) {
        // interaction branch
        layernorm_f16<<<MM, DD, 0, stream>>>(x, g1, b1, yf16);
        launch_gemm<DD>(yf16, WqT, nullptr, qsB, nullptr, MM, NN, /*relu*/1, stream);
        launch_gemm<DD>(yf16, WvT, nullptr, vsB, nullptr, MM, NN, /*relu*/1, stream);

        sig_kernel<<<(BB * NN + 255) / 256, 256, 0, stream>>>(qsB, sig);
        metric_l1<<<(BB * (NN / 2) + 255) / 256, 256, 0, stream>>>(sig, mW1, mb1, hid);
        metric_l2<<<(BB * NN + 255) / 256, 256, 0, stream>>>(hid, mW2, mb2, base_m, metric);

        window_attn<<<BB * TT * HH, HDIM, 0, stream>>>(qsB, vsB, metric, hebbian, gated);
        launch_gemm<NN>(gated, decT, nullptr, decOut, nullptr, MM, DD, /*none*/0, stream);

        // feed-forward branch
        layernorm_f16<<<MM, DD, 0, stream>>>(x, g2, b2, yf16);
        launch_gemm<DD>(yf16, fW1T, fb1, nullptr, h1, MM, FFD, /*gelu*/2, stream);
        launch_gemm<FFD>(h1, fW2T, fb2, ffnOut, nullptr, MM, DD, /*none*/0, stream);

        residual_add<<<(MM * DD + 255) / 256, 256, 0, stream>>>(x, decOut, ffnOut, MM * DD);
    }

    // --- final LN + vocab projection ---
    layernorm_f16<<<MM, DD, 0, stream>>>(x, gf, bf, yf16);
    launch_gemm<DD>(yf16, WlT, bl, out, nullptr, MM, VOC, /*none*/0, stream);
}
